// Decoder_76029511074200
// MI455X (gfx1250) — matrix operand order repaired and checked by reading, not compiled, not run on hardware
//
#include <hip/hip_runtime.h>

// ---------------- problem constants ----------------
#define PB     256
#define PL     128
#define PV     4096
#define EENC   1024
#define DDEC   1024
#define EEMB   512

typedef __attribute__((ext_vector_type(16))) _Float16 v16h;
typedef __attribute__((ext_vector_type(8)))  _Float16 v8h;
typedef __attribute__((ext_vector_type(4)))  _Float16 v4h;
typedef __attribute__((ext_vector_type(8)))  float    v8f;
typedef __attribute__((ext_vector_type(4)))  float    v4f;

union V16U { v16h v; v8h h[2]; };

static __device__ __forceinline__ v8f wmma_f16(const v16h& a, const v16h& b, const v8f& c) {
    return __builtin_amdgcn_wmma_f32_16x16x32_f16(false, a, false, b, (short)0, c, false, false);
}

// generic (shared) pointer -> 32-bit LDS offset for async-to-LDS ops
static __device__ __forceinline__ unsigned lds_off(const void* p) {
    return (unsigned)(unsigned long long)(__attribute__((address_space(3))) const void*)p;
}

// async global->LDS 16-byte copy (per lane), tracked with ASYNCcnt
static __device__ __forceinline__ void async_g2l_b128(unsigned lds_byte_off, const void* gsrc) {
    unsigned long long ga = (unsigned long long)gsrc;
    asm volatile("global_load_async_to_lds_b128 %0, %1, off"
                 :: "v"(lds_byte_off), "v"(ga) : "memory");
}
static __device__ __forceinline__ void wait_async0() {
    asm volatile("s_wait_asynccnt 0x0" ::: "memory");
}

// ---------------- f32 -> f16 convert ----------------
__global__ void cvt_f16_kernel(const float* __restrict__ s, _Float16* __restrict__ d, int n) {
    int i = blockIdx.x * blockDim.x + threadIdx.x;
    if (i < n) d[i] = (_Float16)s[i];
}

// biasq = W1_b + W2_b
__global__ void bias_sum_kernel(const float* __restrict__ a, const float* __restrict__ b,
                                float* __restrict__ o, int n) {
    int i = blockIdx.x * blockDim.x + threadIdx.x;
    if (i < n) o[i] = a[i] + b[i];
}

// ---------------- generic GEMM: C = act(A @ W^T + bias) ----------------
// A: M x K f32 (lda), W: N x K f16 row-major, C: f32 with ldc and column offset.
// Block = 256 threads = 8 waves; each wave owns one 16x16 output tile.
__global__ __launch_bounds__(256) void gemm_f16_kernel(
    const float* __restrict__ A, int lda,
    const _Float16* __restrict__ W,
    const float* __restrict__ bias,
    float* __restrict__ C, int ldc, int ccol,
    int M, int N, int K, int act)
{
    int lane = threadIdx.x & 31;
    int wave = threadIdx.x >> 5;
    int n0 = (blockIdx.x * 8 + wave) * 16;
    int m0 = blockIdx.y * 16;
    if (n0 >= N || m0 >= M) return;              // uniform per wave

    int hi = lane >> 4;                           // 0: low half, 1: high half
    int lq = lane & 15;
    const float*    arow = A + (size_t)(m0 + lq) * lda;
    const _Float16* wrow = W + (size_t)(n0 + lq) * K;

    v8f acc = {};
    for (int k0 = 0; k0 < K; k0 += 32) {
        // A fragment: per 16-bit 16x32 layout, lane half 'hi' holds K runs
        // [k0+8*hi .. +7] and [k0+16+8*hi .. +7]
        V16U a;
        const float* p1 = arow + k0 + 8 * hi;
        const float* p2 = arow + k0 + 16 + 8 * hi;
        v4f f0 = *(const v4f*)p1;      v4f f1 = *(const v4f*)(p1 + 4);
        v4f f2 = *(const v4f*)p2;      v4f f3 = *(const v4f*)(p2 + 4);
#pragma unroll
        for (int i = 0; i < 4; ++i) {
            a.v[i]      = (_Float16)f0[i];
            a.v[4 + i]  = (_Float16)f1[i];
            a.v[8 + i]  = (_Float16)f2[i];
            a.v[12 + i] = (_Float16)f3[i];
        }
        // B fragment: lane = column n, halves cover K [k0+8*hi .. +7] and [k0+16+8*hi .. +7]
        V16U b;
        const _Float16* pb = wrow + k0 + 8 * hi;
        b.h[0] = *(const v8h*)pb;
        b.h[1] = *(const v8h*)(pb + 16);

        acc = wmma_f16(a.v, b.v, acc);
    }

    int n = n0 + lq;
    float bv = bias ? bias[n] : 0.0f;
#pragma unroll
    for (int j = 0; j < 8; ++j) {
        int m = m0 + j + 8 * hi;
        float v = acc[j] + bv;
        if (act == 1) v = tanhf(v);
        C[(size_t)m * ldc + ccol + n] = v;
    }
}

// ---------------- fused Bahdanau attention ----------------
// One workgroup per batch element b. enc[b] (128x1024) staged as f16 in LDS.
// logit[l] = sum_d tanh( (enc @ W1^T)[l,d] + q[b,d] ) * Vw[d]; softmax over l;
// context[e] = sum_l attn[l]*enc[b,l,e]  (re-read f32 from L2 for accuracy).
#define ENC_STRIDE 1032   // 1024 + 8 halves padding

__global__ __launch_bounds__(256) void attn_kernel(
    const float* __restrict__ enc,        // B,L,EENC f32
    const _Float16* __restrict__ W1h,     // DDEC x EENC f16
    const float* __restrict__ q,          // B x DDEC  (h@W2^T + W1_b + W2_b)
    const float* __restrict__ Vw,         // DDEC
    float* __restrict__ attn_out,         // B x L
    float* __restrict__ ctx_out,          // B x ldctx (cols 0..1023)
    int ldctx)
{
    __shared__ __align__(16) _Float16 encs[PL * ENC_STRIDE];    // 264192 B
    __shared__ __align__(16) _Float16 w1s[16 * ENC_STRIDE];     //  33024 B
    __shared__ float logit[PL];
    __shared__ float red[PL];
    __shared__ float attn_s[PL];
    __shared__ float vws[DDEC];

    int b   = blockIdx.x;
    int tid = threadIdx.x;
    const float* encb = enc + (size_t)b * PL * EENC;

    // stage enc -> f16 LDS (coalesced float4 loads, converted in VGPRs)
    for (int i = tid; i < PL * EENC / 4; i += 256) {
        int e4  = i * 4;
        int row = e4 >> 10, col = e4 & 1023;
        v4f f = *(const v4f*)(encb + e4);
        v4h o;
        o[0] = (_Float16)f[0]; o[1] = (_Float16)f[1];
        o[2] = (_Float16)f[2]; o[3] = (_Float16)f[3];
        *(v4h*)&encs[row * ENC_STRIDE + col] = o;
    }
    for (int i = tid; i < DDEC; i += 256) vws[i] = Vw[i];
    if (tid < PL) logit[tid] = 0.0f;

    int lane = tid & 31, wave = tid >> 5;
    int hi = lane >> 4, lq = lane & 15;
    int m0 = wave * 16;                              // 8 waves cover l = 0..127
    const _Float16* ab = &encs[(m0 + lq) * ENC_STRIDE];
    const _Float16* bb = &w1s[lq * ENC_STRIDE];

    for (int nc = 0; nc < DDEC / 16; ++nc) {
        __syncthreads();                              // prior reads of w1s done
        // stage W1 rows [nc*16, nc*16+16): async global->LDS, no VGPR roundtrip
        for (int i = tid; i < 2048; i += 256) {       // 2048 x 16B segments
            int h0 = i * 8;
            int r = h0 >> 10, c = h0 & 1023;
            async_g2l_b128(lds_off(&w1s[r * ENC_STRIDE + c]),
                           W1h + (size_t)(nc * 16 + r) * EENC + c);
        }
        wait_async0();
        __syncthreads();

        // prefetch next W1 tile toward L2/L0 while this chunk computes
        if (nc + 1 < DDEC / 16 && tid < 16)
            __builtin_prefetch(W1h + (size_t)((nc + 1) * 16 + tid) * EENC, 0, 1);

        v8f acc = {};
        for (int k0 = 0; k0 < EENC; k0 += 32) {
            V16U a, bf;
            a.h[0]  = *(const v8h*)(ab + k0 + 8 * hi);
            a.h[1]  = *(const v8h*)(ab + k0 + 16 + 8 * hi);
            bf.h[0] = *(const v8h*)(bb + k0 + 8 * hi);
            bf.h[1] = *(const v8h*)(bb + k0 + 16 + 8 * hi);
            acc = wmma_f16(a.v, bf.v, acc);
        }

        int   d  = nc * 16 + lq;
        float qv = q[b * DDEC + d];
        float vw = vws[d];
#pragma unroll
        for (int j = 0; j < 8; ++j) {
            float p = tanhf(acc[j] + qv) * vw;
            // reduce over the 16 d-lanes of this half-wave
            p += __shfl_xor(p, 1);
            p += __shfl_xor(p, 2);
            p += __shfl_xor(p, 4);
            p += __shfl_xor(p, 8);
            if (lq == 0) atomicAdd(&logit[m0 + j + 8 * hi], p);
        }
    }
    __syncthreads();

    // softmax over L = 128 (V_b is a uniform shift -> dropped)
    if (tid < PL) red[tid] = logit[tid];
    __syncthreads();
    for (int s = 64; s > 0; s >>= 1) {
        if (tid < s) red[tid] = fmaxf(red[tid], red[tid + s]);
        __syncthreads();
    }
    float mx = red[0];
    __syncthreads();
    if (tid < PL) { float e = __expf(logit[tid] - mx); attn_s[tid] = e; red[tid] = e; }
    __syncthreads();
    for (int s = 64; s > 0; s >>= 1) {
        if (tid < s) red[tid] += red[tid + s];
        __syncthreads();
    }
    float inv = 1.0f / red[0];
    __syncthreads();
    if (tid < PL) {
        attn_s[tid] *= inv;
        attn_out[b * PL + tid] = attn_s[tid];
    }
    __syncthreads();

    // context: thread owns e = tid + 256*i, full f32 re-read (coalesced rows)
    float c0 = 0.f, c1 = 0.f, c2 = 0.f, c3 = 0.f;
    for (int l = 0; l < PL; ++l) {
        float a = attn_s[l];
        const float* r = encb + (size_t)l * EENC;
        c0 += a * r[tid];
        c1 += a * r[tid + 256];
        c2 += a * r[tid + 512];
        c3 += a * r[tid + 768];
    }
    float* cr = ctx_out + (size_t)b * ldctx;
    cr[tid] = c0; cr[tid + 256] = c1; cr[tid + 512] = c2; cr[tid + 768] = c3;
}

// ---------------- GRU gates (PyTorch r,z,n layout) ----------------
__global__ void gru_gates_kernel(const float* __restrict__ gi, const float* __restrict__ gh,
                                 const float* __restrict__ h0, float* __restrict__ hnew) {
    int i = blockIdx.x * blockDim.x + threadIdx.x;     // B*DDEC
    if (i >= PB * DDEC) return;
    int bb = i >> 10, d = i & 1023;
    const float* gib = gi + (size_t)bb * 3 * DDEC;
    const float* ghb = gh + (size_t)bb * 3 * DDEC;
    float r = 1.f / (1.f + __expf(-(gib[d]            + ghb[d])));
    float z = 1.f / (1.f + __expf(-(gib[DDEC + d]     + ghb[DDEC + d])));
    float n = tanhf(gib[2 * DDEC + d] + r * ghb[2 * DDEC + d]);
    hnew[i] = (1.f - z) * n + z * h0[i];
}

// ---------------- host launcher ----------------
static inline size_t alignup(size_t x) { return (x + 255) & ~(size_t)255; }

extern "C" void kernel_launch(void* const* d_in, const int* in_sizes, int n_in,
                              void* d_out, int out_size, void* d_ws, size_t ws_size,
                              hipStream_t stream) {
    const float* x        = (const float*)d_in[0];   // B,1,V
    const float* hidden   = (const float*)d_in[1];   // 1,B,DDEC
    const float* enc      = (const float*)d_in[2];   // B,L,EENC
    const float* W1_w     = (const float*)d_in[3];
    const float* W1_b     = (const float*)d_in[4];
    const float* W2_w     = (const float*)d_in[5];
    const float* W2_b     = (const float*)d_in[6];
    const float* V_w      = (const float*)d_in[7];
    // d_in[8] = V_b : scalar shift, softmax-invariant -> unused
    const float* o2e_w    = (const float*)d_in[9];
    const float* o2e_b    = (const float*)d_in[10];
    const float* gru_wi   = (const float*)d_in[11];
    const float* gru_wh   = (const float*)d_in[12];
    const float* gru_bi   = (const float*)d_in[13];
    const float* gru_bh   = (const float*)d_in[14];
    const float* fc1_w    = (const float*)d_in[15];
    const float* fc1_b    = (const float*)d_in[16];
    const float* fc2_w    = (const float*)d_in[17];
    const float* fc2_b    = (const float*)d_in[18];

    float* out_logits = (float*)d_out;                         // B*V
    float* out_hnew   = out_logits + (size_t)PB * PV;          // B*DDEC
    float* out_attn   = out_hnew + (size_t)PB * DDEC;          // B*L

    // workspace layout
    char* ws = (char*)d_ws;  size_t off = 0;
    auto take = [&](size_t bytes) { char* p = ws + off; off = alignup(off + bytes); return p; };
    _Float16* W1h   = (_Float16*)take((size_t)DDEC * EENC * 2);
    _Float16* W2h   = (_Float16*)take((size_t)DDEC * DDEC * 2);
    _Float16* o2eh  = (_Float16*)take((size_t)EEMB * PV * 2);
    _Float16* gwih  = (_Float16*)take((size_t)3 * DDEC * (EEMB + EENC) * 2);
    _Float16* gwhh  = (_Float16*)take((size_t)3 * DDEC * DDEC * 2);
    _Float16* fc1h  = (_Float16*)take((size_t)EEMB * DDEC * 2);
    _Float16* fc2h  = (_Float16*)take((size_t)PV * EEMB * 2);
    float*    biasq = (float*)take((size_t)DDEC * 4);
    float*    qbuf  = (float*)take((size_t)PB * DDEC * 4);
    float*    rnnin = (float*)take((size_t)PB * (EENC + EEMB) * 4);   // [ctx | x_emb]
    float*    gibuf = (float*)take((size_t)PB * 3 * DDEC * 4);
    float*    ghbuf = (float*)take((size_t)PB * 3 * DDEC * 4);
    float*    t1buf = (float*)take((size_t)PB * EEMB * 4);
    (void)ws_size; (void)n_in; (void)in_sizes; (void)out_size;

    auto cvt = [&](const float* s, _Float16* d, int n) {
        cvt_f16_kernel<<<(n + 255) / 256, 256, 0, stream>>>(s, d, n);
    };
    cvt(W1_w,   W1h,  DDEC * EENC);
    cvt(W2_w,   W2h,  DDEC * DDEC);
    cvt(o2e_w,  o2eh, EEMB * PV);
    cvt(gru_wi, gwih, 3 * DDEC * (EEMB + EENC));
    cvt(gru_wh, gwhh, 3 * DDEC * DDEC);
    cvt(fc1_w,  fc1h, EEMB * DDEC);
    cvt(fc2_w,  fc2h, PV * EEMB);
    bias_sum_kernel<<<(DDEC + 255) / 256, 256, 0, stream>>>(W1_b, W2_b, biasq, DDEC);

    const int KIN = EENC + EEMB;  // 1536

    // q = hidden @ W2^T + (W1_b + W2_b)
    gemm_f16_kernel<<<dim3(DDEC / 128, PB / 16), 256, 0, stream>>>(
        hidden, DDEC, W2h, biasq, qbuf, DDEC, 0, PB, DDEC, DDEC, 0);

    // x_emb = x @ out2emb^T + b  -> rnn_in[:, 1024:1536]
    gemm_f16_kernel<<<dim3(EEMB / 128, PB / 16), 256, 0, stream>>>(
        x, PV, o2eh, o2e_b, rnnin, KIN, EENC, PB, EEMB, PV, 0);

    // fused attention: attn -> d_out, context -> rnn_in[:, 0:1024]
    attn_kernel<<<PB, 256, 0, stream>>>(enc, W1h, qbuf, V_w, out_attn, rnnin, KIN);

    // gi = rnn_in @ gru_wi^T + gru_bi ; gh = hidden @ gru_wh^T + gru_bh
    gemm_f16_kernel<<<dim3(3 * DDEC / 128, PB / 16), 256, 0, stream>>>(
        rnnin, KIN, gwih, gru_bi, gibuf, 3 * DDEC, 0, PB, 3 * DDEC, KIN, 0);
    gemm_f16_kernel<<<dim3(3 * DDEC / 128, PB / 16), 256, 0, stream>>>(
        hidden, DDEC, gwhh, gru_bh, ghbuf, 3 * DDEC, 0, PB, 3 * DDEC, DDEC, 0);

    gru_gates_kernel<<<(PB * DDEC + 255) / 256, 256, 0, stream>>>(gibuf, ghbuf, hidden, out_hnew);

    // t1 = tanh(h_new @ fc1^T + fc1_b) ; out = t1 @ fc2^T + fc2_b
    gemm_f16_kernel<<<dim3(EEMB / 128, PB / 16), 256, 0, stream>>>(
        out_hnew, DDEC, fc1h, fc1_b, t1buf, EEMB, 0, PB, EEMB, DDEC, 1);
    gemm_f16_kernel<<<dim3(PV / 128, PB / 16), 256, 0, stream>>>(
        t1buf, EEMB, fc2h, fc2_b, out_logits, PV, 0, PB, PV, EEMB, 0);
}
